// SAGE_13134009991686
// MI455X (gfx1250) — matrix operand order repaired and checked by reading, not compiled, not run on hardware
//
#include <hip/hip_runtime.h>
#include <hip/hip_bf16.h>
#include <cstdint>

typedef __attribute__((ext_vector_type(16))) _Float16 v16h;
typedef __attribute__((ext_vector_type(8)))  float    v8f;

#define NN 100000
#define NE 1600000
#define NG 128
#define BN_EPS 1e-5f

// ---------------- TDM availability / arity detection -----------------------
#ifdef __has_builtin
#if __has_builtin(__builtin_amdgcn_tensor_load_to_lds)
#define HAVE_TDM 1
#endif
#endif
#ifndef HAVE_TDM
#define HAVE_TDM 0
#endif
#if __has_include(<hip/amd_detail/amd_gfx1250_TDM.h>)
#define TDM_6ARG 1
#else
#define TDM_6ARG 0
#endif

#if HAVE_TDM
typedef __attribute__((ext_vector_type(4))) unsigned tdm_v4u;
typedef __attribute__((ext_vector_type(8))) int      tdm_v8i;
typedef __attribute__((ext_vector_type(4))) int      tdm_v4i;

// Issue a 2D TENSOR_LOAD_TO_LDS: tile of [tile_rows x width] f32 starting at g,
// row stride = width. tensor_dim1 = rows_avail so the TDM zero-fills rows past
// the end of the node array. lds_addr: low 32 bits of a generic LDS pointer are
// the LDS byte offset (aperture bits live in [63:32]).
__device__ __forceinline__ void tdm_load_2d(const float* g, unsigned lds_addr,
                                            int width, int rows_avail, int tile_rows) {
  unsigned long long ga = (unsigned long long)(uintptr_t)g;
  tdm_v4u g0;
  g0[0] = 1u;                                               // count=1, user desc
  g0[1] = lds_addr;                                         // lds_addr [63:32]
  g0[2] = (unsigned)ga;                                     // global_addr lo
  g0[3] = (unsigned)((ga >> 32) & 0x1FFFFFFu) | (2u << 30); // addr hi | type=2
  tdm_v8i g1;
  g1[0] = (int)(2u << 16);                                  // data_size=4B
  g1[1] = (int)(((unsigned)width & 0xFFFFu) << 16);         // tensor_dim0 lo16
  g1[2] = (int)((((unsigned)width >> 16) & 0xFFFFu) |
                (((unsigned)rows_avail & 0xFFFFu) << 16));  // dim0 hi | dim1 lo
  g1[3] = (int)((((unsigned)rows_avail >> 16) & 0xFFFFu) |
                (((unsigned)width & 0xFFFFu) << 16));       // dim1 hi | tile_dim0
  g1[4] = (int)((unsigned)tile_rows & 0xFFFFu);             // tile_dim1
  g1[5] = width;                                            // tensor_dim0_stride
  g1[6] = 0;
  g1[7] = 0;
  tdm_v4i z = {0, 0, 0, 0};
#if TDM_6ARG
  tdm_v8i z8 = {0, 0, 0, 0, 0, 0, 0, 0};
  __builtin_amdgcn_tensor_load_to_lds(g0, g1, z, z, z8, 0);
#else
  __builtin_amdgcn_tensor_load_to_lds(g0, g1, z, z, 0);
#endif
}
#endif  // HAVE_TDM

// ---------------- WMMA fragment builders -----------------------------------
// A (f16 16x32): lane<16 -> M=lane, K={0..7,16..23}; lane>=16 -> K={8..15,24..31}
// Virtual A row = [ s_row * rc (DIN floats) | x_row (DIN floats) ], K = 2*DIN.
template <int DIN>
__device__ __forceinline__ v16h make_a_cat(const float* __restrict__ sr,
                                           const float* __restrict__ xr,
                                           float rc, int lane, int k0) {
  const int koff = (lane < 16) ? 0 : 8;
  v16h a;
#pragma unroll
  for (int e = 0; e < 16; ++e) {
    int kk = k0 + koff + ((e < 8) ? e : e + 8);
    float v = (kk < DIN) ? sr[kk] * rc : xr[kk - DIN];
    a[e] = (_Float16)v;
  }
  return a;
}
// Plain f32 A row (head GEMM)
__device__ __forceinline__ v16h make_a(const float* __restrict__ arow, int lane, int k0) {
  const int koff = (lane < 16) ? 0 : 8;
  const float* p = arow + k0 + koff;
  v16h a;
#pragma unroll
  for (int e = 0; e < 16; ++e) a[e] = (_Float16)p[(e < 8) ? e : e + 8];
  return a;
}
// B (f16 32x16): lane -> N=lane&15, K = {0..7,16..23} (lane<16) / {8..15,24..31} (lane>=16)
__device__ __forceinline__ v16h make_b(const _Float16* __restrict__ w, int dout,
                                       int n, int lane, int k0) {
  const int kboff = (lane < 16) ? 0 : 8;
  const _Float16* p = w + (long)(k0 + kboff) * dout + n;
  v16h b;
#pragma unroll
  for (int e = 0; e < 16; ++e) b[e] = p[((e < 8) ? e : e + 8) * dout];
  return b;
}
// f32-weight variant (head GEMM)
__device__ __forceinline__ v16h make_bf(const float* __restrict__ w, int dout,
                                        int n, int lane, int k0) {
  const int kboff = (lane < 16) ? 0 : 8;
  const float* p = w + (long)(k0 + kboff) * dout + n;
  v16h b;
#pragma unroll
  for (int e = 0; e < 16; ++e) b[e] = (_Float16)p[((e < 8) ? e : e + 8) * dout];
  return b;
}

// ---------------- weight pack: wcat = f16([wl ; wr]) ------------------------
__global__ void pack_weights_kernel(const float* __restrict__ wl,
                                    const float* __restrict__ wr,
                                    _Float16* __restrict__ wcat, int half_elems) {
  int i = blockIdx.x * blockDim.x + threadIdx.x;
  if (i >= 2 * half_elems) return;
  wcat[i] = (_Float16)((i < half_elems) ? wl[i] : wr[i - half_elems]);
}

// ---------------- degree ----------------------------------------------------
__global__ void deg_kernel(const int* __restrict__ dst, float* __restrict__ cnt) {
  int e = blockIdx.x * blockDim.x + threadIdx.x;
  if (e < NE) atomicAdd(&cnt[dst[e]], 1.0f);
}

// ---------------- edge scatter-add: s[dst] += x[src] ------------------------
template <int D>
__global__ void scatter_kernel(const int* __restrict__ src, const int* __restrict__ dst,
                               const float* __restrict__ x, float* __restrict__ s) {
  constexpr int CH = D / 4;
  long tid = (long)blockIdx.x * blockDim.x + threadIdx.x;
  if (tid >= (long)NE * CH) return;
  int e = (int)(tid / CH);
  int c = (int)(tid % CH);
  int sn = src[e], dn = dst[e];
  const float4 v = *(const float4*)(x + (long)sn * D + c * 4);
  float* p = s + (long)dn * D + c * 4;
  atomicAdd(p + 0, v.x); atomicAdd(p + 1, v.y);
  atomicAdd(p + 2, v.z); atomicAdd(p + 3, v.w);
}

// ---------------- SAGEConv dense: out = (s/cnt)@wl + b + x@wr ---------------
// Block: MT m-tiles (MT*16 nodes) x DOUT cols, 8 waves; wave does NTPW n-tiles.
// A tiles staged in LDS once per block via TDM (or cooperative fallback).
// Single contraction over K = 2*DIN against packed wcat = [wl ; wr].
template <int DIN, int DOUT, int MT, int NTPW>
__global__ __launch_bounds__(256) void sage_wmma_kernel(
    const float* __restrict__ s, const float* __restrict__ cnt,
    const float* __restrict__ x, const _Float16* __restrict__ wcat,
    const float* __restrict__ bias, float* __restrict__ out) {
  constexpr int WPM = (DOUT / 16) / NTPW;   // waves per m-tile
  static_assert(MT * WPM == 8, "8 waves per block");
  constexpr int TILEF = MT * 16 * DIN;      // floats per operand tile
  __shared__ float smem[TILEF * 2];
  const int row_base = blockIdx.x * (MT * 16);

#if HAVE_TDM
  if (threadIdx.x < 32) {                   // wave 0 drives both DMAs
    tdm_load_2d(s + (long)row_base * DIN, (unsigned)(uintptr_t)&smem[0],
                DIN, NN - row_base, MT * 16);
    tdm_load_2d(x + (long)row_base * DIN, (unsigned)(uintptr_t)&smem[TILEF],
                DIN, NN - row_base, MT * 16);
    __builtin_amdgcn_s_wait_tensorcnt(0);
  }
#else
  for (int i = threadIdx.x; i < TILEF; i += 256) {
    int r = i / DIN, cc = i - r * DIN;
    int grow = row_base + r;
    float vs = 0.f, vx = 0.f;
    if (grow < NN) {
      vs = s[(long)grow * DIN + cc];
      vx = x[(long)grow * DIN + cc];
    }
    smem[i] = vs;
    smem[TILEF + i] = vx;
  }
#endif
  __syncthreads();

  const int wid = threadIdx.x >> 5;
  const int lane = threadIdx.x & 31;
  const int m_local = wid / WPM;
  const int n_group = wid - m_local * WPM;
  if (row_base + m_local * 16 >= NN) return;           // wave-uniform
  const int grow = row_base + m_local * 16 + (lane & 15);
  const float rc = 1.0f / fmaxf(cnt[grow], 1.0f);
  const float* sr = smem + (m_local * 16 + (lane & 15)) * DIN;
  const float* xr = sr + TILEF;

  v8f acc[NTPW];
#pragma unroll
  for (int t = 0; t < NTPW; ++t) acc[t] = (v8f){};

#pragma unroll
  for (int k0 = 0; k0 < 2 * DIN; k0 += 32) {
    v16h a = make_a_cat<DIN>(sr, xr, rc, lane, k0);
#pragma unroll
    for (int t = 0; t < NTPW; ++t) {
      int n = (n_group * NTPW + t) * 16 + (lane & 15);
      v16h b = make_b(wcat, DOUT, n, lane, k0);
      acc[t] = __builtin_amdgcn_wmma_f32_16x16x32_f16(false, a, false, b, (short)0,
                                                      acc[t], false, false);
    }
  }

#pragma unroll
  for (int t = 0; t < NTPW; ++t) {
    int n = (n_group * NTPW + t) * 16 + (lane & 15);
    float bb = bias[n];
#pragma unroll
    for (int r = 0; r < 8; ++r) {
      int row = row_base + m_local * 16 + r + ((lane < 16) ? 0 : 8);
      out[(long)row * DOUT + n] = acc[t][r] + bb;
    }
  }
}

// ---------------- BN statistics ---------------------------------------------
__global__ __launch_bounds__(256) void bn_stats_kernel(const float* __restrict__ h,
                                                       float* __restrict__ stats, int D) {
  int f = blockIdx.x;
  float sm = 0.f, sq = 0.f;
  for (int nnode = threadIdx.x; nnode < NN; nnode += 256) {
    float v = h[(long)nnode * D + f];
    sm += v; sq += v * v;
  }
  __shared__ float sh[256], sh2[256];
  sh[threadIdx.x] = sm; sh2[threadIdx.x] = sq;
  __syncthreads();
  for (int o = 128; o > 0; o >>= 1) {
    if (threadIdx.x < o) { sh[threadIdx.x] += sh[threadIdx.x + o];
                           sh2[threadIdx.x] += sh2[threadIdx.x + o]; }
    __syncthreads();
  }
  if (threadIdx.x == 0) {
    float m = sh[0] / (float)NN;
    stats[f] = m;
    stats[D + f] = sh2[0] / (float)NN - m * m;   // biased variance
  }
}

// ---------------- BN apply + ReLU -------------------------------------------
__global__ void bn_relu_kernel(float* __restrict__ h, const float* __restrict__ stats,
                               const float* __restrict__ g, const float* __restrict__ be,
                               int D) {
  long tid = (long)blockIdx.x * blockDim.x + threadIdx.x;
  if (tid >= (long)NN * D) return;
  int f = (int)(tid % D);
  float sc = g[f] * rsqrtf(stats[D + f] + BN_EPS);
  float v = (h[tid] - stats[f]) * sc + be[f];
  h[tid] = fmaxf(v, 0.f);
}

// ---------------- per-graph max pool (values >= 0) --------------------------
__global__ void pool_kernel(const float* __restrict__ h, const int* __restrict__ batch,
                            float* __restrict__ pooled) {
  long tid = (long)blockIdx.x * blockDim.x + threadIdx.x;
  if (tid >= (long)NN * 512) return;
  int nnode = (int)(tid >> 9);
  int f = (int)(tid & 511);
  atomicMax((int*)&pooled[(long)batch[nnode] * 512 + f], __float_as_int(h[tid]));
}

// ---------------- generic WMMA GEMM (head): out = relu(A@W + b) -------------
__global__ __launch_bounds__(256) void gemm_wmma_kernel(
    const float* __restrict__ A, const float* __restrict__ W,
    const float* __restrict__ bias, float* __restrict__ out,
    int M, int K, int Nout, int relu) {
  int wave = (blockIdx.x * 256 + threadIdx.x) >> 5;
  int lane = threadIdx.x & 31;
  int nt = Nout >> 4;
  int m_tile = wave / nt;
  int n_tile = wave - m_tile * nt;
  if (m_tile * 16 >= M) return;
  int mrow = m_tile * 16 + (lane & 15);
  int n = n_tile * 16 + (lane & 15);
  const float* arow = A + (long)mrow * K;
  v8f c = {};
  for (int k0 = 0; k0 < K; k0 += 32) {
    v16h a = make_a(arow, lane, k0);
    v16h b = make_bf(W, Nout, n, lane, k0);
    c = __builtin_amdgcn_wmma_f32_16x16x32_f16(false, a, false, b, (short)0, c,
                                               false, false);
  }
  float bb = bias[n];
#pragma unroll
  for (int r = 0; r < 8; ++r) {
    int row = m_tile * 16 + r + ((lane < 16) ? 0 : 8);
    float v = c[r] + bb;
    if (relu) v = fmaxf(v, 0.f);
    out[(long)row * Nout + n] = v;
  }
}

// ---------------- final 256 -> 10 head --------------------------------------
__global__ void head2_kernel(const float* __restrict__ hid, const float* __restrict__ w,
                             const float* __restrict__ b, float* __restrict__ out) {
  int tid = blockIdx.x * blockDim.x + threadIdx.x;
  if (tid >= NG * 10) return;
  int g = tid / 10, o = tid - g * 10;
  float acc = b[o];
  for (int k = 0; k < 256; ++k) acc += hid[g * 256 + k] * w[k * 10 + o];
  out[tid] = acc;
}

// ---------------------------------------------------------------------------
extern "C" void kernel_launch(void* const* d_in, const int* in_sizes, int n_in,
                              void* d_out, int out_size, void* d_ws, size_t ws_size,
                              hipStream_t stream) {
  (void)in_sizes; (void)n_in; (void)out_size; (void)ws_size;
  const float* x   = (const float*)d_in[0];
  const int*   ei  = (const int*)d_in[1];
  const int*   bat = (const int*)d_in[2];
  const float* w1l = (const float*)d_in[3];  const float* b1 = (const float*)d_in[4];
  const float* w1r = (const float*)d_in[5];
  const float* w2l = (const float*)d_in[6];  const float* b2 = (const float*)d_in[7];
  const float* w2r = (const float*)d_in[8];
  const float* w3l = (const float*)d_in[9];  const float* b3 = (const float*)d_in[10];
  const float* w3r = (const float*)d_in[11];
  const float* g1  = (const float*)d_in[12]; const float* be1 = (const float*)d_in[13];
  const float* g2  = (const float*)d_in[14]; const float* be2 = (const float*)d_in[15];
  const float* g3  = (const float*)d_in[16]; const float* be3 = (const float*)d_in[17];
  const float* wl1 = (const float*)d_in[18]; const float* bl1 = (const float*)d_in[19];
  const float* wl2 = (const float*)d_in[20]; const float* bl2 = (const float*)d_in[21];
  const int* src = ei;
  const int* dst = ei + NE;

  // workspace layout (floats)
  float* ws     = (float*)d_ws;
  float* cnt    = ws;                       // 100096
  float* sbuf   = cnt  + 100096;            // NN*64
  float* h1     = sbuf + 6400000;           // NN*16
  float* h2     = h1   + 1600000;           // NN*64
  float* h3     = h2   + 6400000;           // NN*512
  float* stats  = h3   + 51200000;          // 1024
  float* pooled = stats + 1024;             // NG*512
  float* hid    = pooled + 65536;           // NG*256
  _Float16* wcat = (_Float16*)(hid + 32768); // up to 128*512 halves (128 KB)

  // degrees
  hipMemsetAsync(cnt, 0, NN * sizeof(float), stream);
  deg_kernel<<<(NE + 255) / 256, 256, 0, stream>>>(dst, cnt);

  // ---- layer 1: 32 -> 16 ----
  hipMemsetAsync(sbuf, 0, (size_t)NN * 32 * sizeof(float), stream);
  scatter_kernel<32><<<(NE * 8 + 255) / 256, 256, 0, stream>>>(src, dst, x, sbuf);
  pack_weights_kernel<<<(2 * 32 * 16 + 255) / 256, 256, 0, stream>>>(w1l, w1r, wcat, 32 * 16);
  sage_wmma_kernel<32, 16, 8, 1><<<(6250 + 7) / 8, 256, 0, stream>>>(
      sbuf, cnt, x, wcat, b1, h1);
  bn_stats_kernel<<<16, 256, 0, stream>>>(h1, stats, 16);
  bn_relu_kernel<<<(int)(((long)NN * 16 + 255) / 256), 256, 0, stream>>>(h1, stats, g1, be1, 16);

  // ---- layer 2: 16 -> 64 ----
  hipMemsetAsync(sbuf, 0, (size_t)NN * 16 * sizeof(float), stream);
  scatter_kernel<16><<<(NE * 4 + 255) / 256, 256, 0, stream>>>(src, dst, h1, sbuf);
  pack_weights_kernel<<<(2 * 16 * 64 + 255) / 256, 256, 0, stream>>>(w2l, w2r, wcat, 16 * 64);
  sage_wmma_kernel<16, 64, 4, 2><<<(6250 + 3) / 4, 256, 0, stream>>>(
      sbuf, cnt, h1, wcat, b2, h2);
  bn_stats_kernel<<<64, 256, 0, stream>>>(h2, stats, 64);
  bn_relu_kernel<<<(int)(((long)NN * 64 + 255) / 256), 256, 0, stream>>>(h2, stats, g2, be2, 64);

  // ---- layer 3: 64 -> 512 ----
  hipMemsetAsync(sbuf, 0, (size_t)NN * 64 * sizeof(float), stream);
  scatter_kernel<64><<<(int)(((long)NE * 16 + 255) / 256), 256, 0, stream>>>(src, dst, h2, sbuf);
  pack_weights_kernel<<<(2 * 64 * 512 + 255) / 256, 256, 0, stream>>>(w3l, w3r, wcat, 64 * 512);
  sage_wmma_kernel<64, 512, 1, 4><<<6250, 256, 0, stream>>>(
      sbuf, cnt, h2, wcat, b3, h3);
  bn_stats_kernel<<<512, 256, 0, stream>>>(h3, stats, 512);
  bn_relu_kernel<<<(int)(((long)NN * 512 + 255) / 256), 256, 0, stream>>>(h3, stats, g3, be3, 512);

  // ---- global max pool ----
  hipMemsetAsync(pooled, 0, (size_t)NG * 512 * sizeof(float), stream);
  pool_kernel<<<(int)(((long)NN * 512 + 255) / 256), 256, 0, stream>>>(h3, bat, pooled);

  // ---- head ----
  gemm_wmma_kernel<<<((NG / 16) * (256 / 16) + 7) / 8, 256, 0, stream>>>(
      pooled, wl1, bl1, hid, NG, 512, 256, 1);
  head2_kernel<<<(NG * 10 + 255) / 256, 256, 0, stream>>>(hid, wl2, bl2, (float*)d_out);
}
